// Quantize_43645457662413
// MI455X (gfx1250) — matrix operand order repaired and checked by reading, not compiled, not run on hardware
//
#include <hip/hip_runtime.h>

// ---------------- types ----------------
typedef __attribute__((ext_vector_type(16))) __bf16 v16bf;
typedef __attribute__((ext_vector_type(8)))  __bf16 bf16x8;   // 16 bytes
typedef __attribute__((ext_vector_type(4)))  __bf16 bf16x4;   // 8 bytes
typedef __attribute__((ext_vector_type(8)))  float  v8f;

#define DIM    256
#define NCODE  1024
#define NTOK   65536
#define QELEMS (NTOK * DIM)          // 16,777,216 quantize elements

// ---------------- workspace layout (bytes) ----------------
#define WS_NORMS 0                               // 1024 f32          (4 KB)
#define WS_EMBT  4096                            // 262144 f32        (1 MB)
#define WS_BFRAG (4096 + 1048576)                // 262144 bf16       (512 KB)
#define WS_IDX   (4096 + 1048576 + 524288)       // 65536 int32       (256 KB)

static __device__ __forceinline__ v16bf cat16(bf16x8 lo, bf16x8 hi) {
    return __builtin_shufflevector(lo, hi, 0,1,2,3,4,5,6,7,8,9,10,11,12,13,14,15);
}

// ---------------------------------------------------------------------------
// Kernel 0: per-code HALF squared norms (negated) + transpose codebook to
// [1024][256] f32 for the coalesced gather. nhalf[n] = -0.5*||e_n||^2 so the
// hot kernel can drop it straight into the WMMA accumulator.
// ---------------------------------------------------------------------------
__global__ __launch_bounds__(256) void vq_prep_codes(const float* __restrict__ embed,
                                                     float* __restrict__ nhalf,
                                                     float* __restrict__ embT,
                                                     float* __restrict__ diffslot) {
    int n = blockIdx.x * 256 + threadIdx.x;      // 0..1023
    float acc = 0.f;
    for (int d = 0; d < DIM; ++d) {
        float v = embed[d * NCODE + n];
        acc += v * v;
        embT[(size_t)n * DIM + d] = v;
    }
    nhalf[n] = -0.5f * acc;
    if (n == 0) *diffslot = 0.f;                 // zero the diff accumulator
}

// ---------------------------------------------------------------------------
// Kernel 1: pre-pack codebook into per-lane WMMA B-fragment layout (bf16).
// Fragment (nt,kk): lane L holds column n = nt*16 + (L&15),
// K slice = kk*32 + (L>=16 ? 16 : 0) + j, j=0..15  -> 16 bf16 (32B) per lane.
// ---------------------------------------------------------------------------
__global__ __launch_bounds__(256) void vq_prep_bfrag(const float* __restrict__ embed,
                                                     __bf16* __restrict__ bfrag) {
    int tid  = threadIdx.x;
    int lane = tid & 31;
    int wid  = blockIdx.x * 8 + (tid >> 5);      // 0..511
    int nt   = wid >> 3;
    int kk   = wid & 7;
    int sel  = lane >> 4;
    int n    = nt * 16 + (lane & 15);
    __bf16* dst = bfrag + ((size_t)wid * 32 + lane) * 16;
#pragma unroll
    for (int j = 0; j < 16; ++j) {
        int k = kk * 32 + sel * 16 + j;
        dst[j] = (__bf16)embed[(size_t)k * NCODE + n];
    }
}

// ---------------------------------------------------------------------------
// Kernel 2 (hot): distance matmul via v_wmma_f32_16x16x32_bf16 + argmin.
// Block = 128 threads (4 waves), 64 tokens/block, grid = 1024.
// B fragments + (-||e||^2/2) double-buffered in registers at N-tile
// granularity; the norm seeds the accumulator (C_init = -||e||^2/2) so
// argmin(dist) == argmax(C_final) and no score FMA chain is needed.
// ---------------------------------------------------------------------------
__global__ __launch_bounds__(128) void vq_argmin_kernel(const float* __restrict__ in,
                                                        const bf16x8* __restrict__ bfrag,
                                                        const float* __restrict__ nhalf,
                                                        int* __restrict__ idxout) {
    __shared__ __bf16 smem[64 * DIM];            // 32 KB staged tokens (bf16)

    int tid = threadIdx.x;

    // ---- stage 64 tokens: coalesced f32 global loads -> bf16 LDS ----
    const float4* in4 = (const float4*)(in + (size_t)blockIdx.x * 64 * DIM);
#pragma unroll
    for (int it = 0; it < 32; ++it) {
        int lin = it * 128 + tid;                // float4 index
        float4 v = in4[lin];
        bf16x4 b = { (__bf16)v.x, (__bf16)v.y, (__bf16)v.z, (__bf16)v.w };
        *(bf16x4*)(smem + (size_t)lin * 4) = b;
    }
    __syncthreads();

    int lane = tid & 31;
    int wv   = tid >> 5;
    int sel  = lane >> 4;                        // 0: lanes 0-15, 1: lanes 16-31
    int l15  = lane & 15;

    // ---- build A fragments in registers (16x32, contiguous k = 16*sel + j, same order as B) ----
    v16bf a[8];
#pragma unroll
    for (int kk = 0; kk < 8; ++kk) {
        int e0 = (wv * 16 + l15) * DIM + kk * 32 + sel * 16;  // K = e0..e0+15
        bf16x8 lo = *(const bf16x8*)(smem + e0);
        bf16x8 hi = *(const bf16x8*)(smem + e0 + 8);
        a[kk] = cat16(lo, hi);
    }

    float maxv[8];
    int   mini[8];
#pragma unroll
    for (int r = 0; r < 8; ++r) { maxv[r] = -3.4e38f; mini[r] = 0; }

    // Named double buffers (constant indices -> no scratch spills).
#define LOADB(NT, LO, HI, NRM) do {                                           \
        NRM = nhalf[(NT) * 16 + l15];                                         \
        const bf16x8* bp_ = bfrag + ((size_t)(NT) * 256 + lane) * 2;          \
        _Pragma("unroll")                                                     \
        for (int kk = 0; kk < 8; ++kk) {                                      \
            LO[kk] = bp_[kk * 64];                                            \
            HI[kk] = bp_[kk * 64 + 1];                                        \
        }                                                                     \
    } while (0)

#define COMPUTE(NT, LO, HI, NRM) do {                                         \
        v8f c_ = {NRM, NRM, NRM, NRM, NRM, NRM, NRM, NRM};                    \
        _Pragma("unroll")                                                     \
        for (int kk = 0; kk < 8; ++kk) {                                      \
            v16bf bv_ = cat16(LO[kk], HI[kk]);                                \
            c_ = __builtin_amdgcn_wmma_f32_16x16x32_bf16(false, a[kk], false, \
                                    bv_, (short)0, c_, false, false);         \
        }                                                                     \
        _Pragma("unroll")                                                     \
        for (int r = 0; r < 8; ++r) {                                         \
            if (c_[r] > maxv[r]) { maxv[r] = c_[r]; mini[r] = (NT) * 16 + l15; } \
        }                                                                     \
    } while (0)

    bf16x8 b0lo[8], b0hi[8], b1lo[8], b1hi[8];
    float  n0, n1;

    LOADB(0, b0lo, b0hi, n0);
    for (int nt = 0; nt < 64; nt += 2) {
        LOADB(nt + 1, b1lo, b1hi, n1);           // prefetch tile nt+1
        COMPUTE(nt, b0lo, b0hi, n0);             // overlaps with loads above
        if (nt < 62) LOADB(nt + 2, b0lo, b0hi, n0);  // prefetch tile nt+2
        COMPUTE(nt + 1, b1lo, b1hi, n1);
    }
#undef LOADB
#undef COMPUTE

    // ---- argmax reduction (== argmin of distance) across the 16 lanes ----
    for (int m = 1; m < 16; m <<= 1) {
#pragma unroll
        for (int r = 0; r < 8; ++r) {
            float ov = __shfl_xor(maxv[r], m, 32);
            int   oi = __shfl_xor(mini[r], m, 32);
            if (ov > maxv[r] || (ov == maxv[r] && oi < mini[r])) {
                maxv[r] = ov; mini[r] = oi;
            }
        }
    }
    if (l15 == 0) {
        int base = blockIdx.x * 64 + wv * 16 + sel * 8;      // token M = r + 8*sel
#pragma unroll
        for (int r = 0; r < 8; ++r) idxout[base + r] = mini[r];
    }
}

// ---------------------------------------------------------------------------
// Kernel 3: gather quantize rows (coalesced from transposed codebook),
// fused mean-squared-diff reduction, index output (as f32).
// ---------------------------------------------------------------------------
__global__ __launch_bounds__(256) void vq_gather_kernel(const float* __restrict__ in,
                                                        const float* __restrict__ embT,
                                                        const int* __restrict__ idxbuf,
                                                        float* __restrict__ out) {
    __shared__ float red[256];
    int d  = threadIdx.x;
    int t0 = blockIdx.x * 8;
    float acc = 0.f;
#pragma unroll
    for (int i = 0; i < 8; ++i) {
        int t = t0 + i;
        int idx = idxbuf[t];
        float q = embT[(size_t)idx * DIM + d];
        float x = in[(size_t)t * DIM + d];
        out[(size_t)t * DIM + d] = q;            // quantize_st == quantize
        float e = q - x;
        acc += e * e;
        if (d == 0) out[(size_t)QELEMS + 1 + t] = (float)idx;   // embed_ind
    }
    red[d] = acc;
    __syncthreads();
    for (int s = 128; s > 0; s >>= 1) {
        if (d < s) red[d] += red[d + s];
        __syncthreads();
    }
    if (d == 0) atomicAdd(out + QELEMS, red[0] * (1.0f / (float)QELEMS));  // diff
}

// ---------------------------------------------------------------------------
extern "C" void kernel_launch(void* const* d_in, const int* in_sizes, int n_in,
                              void* d_out, int out_size, void* d_ws, size_t ws_size,
                              hipStream_t stream) {
    const float* in    = (const float*)d_in[0];   // [16,64,64,256] f32
    const float* embed = (const float*)d_in[1];   // [256,1024] f32
    float* out = (float*)d_out;

    char* ws      = (char*)d_ws;
    float*  nhalf = (float*)(ws + WS_NORMS);
    float*  embT  = (float*)(ws + WS_EMBT);
    __bf16* bfrag = (__bf16*)(ws + WS_BFRAG);
    int*    idxb  = (int*)(ws + WS_IDX);

    vq_prep_codes <<<4,        256, 0, stream>>>(embed, nhalf, embT, out + QELEMS);
    vq_prep_bfrag <<<64,       256, 0, stream>>>(embed, bfrag);
    vq_argmin_kernel<<<NTOK/64, 128, 0, stream>>>(in, (const bf16x8*)bfrag, nhalf, idxb);
    vq_gather_kernel<<<NTOK/8,  256, 0, stream>>>(in, embT, idxb, out);
}
